// MemoryBank_12352325944207
// MI455X (gfx1250) — compile-verified, matrix operand order checked by reading, not run on hardware
//
#include <hip/hip_runtime.h>
#include <hip/hip_bf16.h>
#include <math.h>

typedef __attribute__((ext_vector_type(16))) _Float16 v16h;
typedef __attribute__((ext_vector_type(8)))  _Float16 v8h;
typedef __attribute__((ext_vector_type(8)))  float    v8f;

#define C_DIM   256
#define N_SLOTS 512
#define HW      4096
#define B_DIM   16
#define MT      32          // tokens per block (2 WMMA M-tiles)
#define SCALE_F 10.0f
#define EPS_F   1e-12f

union FragU { v16h v; v8h h[2]; };

// ---------------------------------------------------------------------------
// Prep: mn_f16[n][c]   = f16( memory[n][c] / max(||memory[n]||, eps) )
//       memT_f16[c][n] = f16( memory[n][c] )          (B operand for GEMM2)
// ---------------------------------------------------------------------------
__global__ void mb_prep_kernel(const float* __restrict__ memory,
                               _Float16* __restrict__ mn,
                               _Float16* __restrict__ memT) {
    __shared__ float red[256];
    const int n = blockIdx.x;
    const int c = threadIdx.x;
    float v = memory[n * C_DIM + c];
    red[c] = v * v;
    __syncthreads();
    for (int s = 128; s > 0; s >>= 1) {
        if (c < s) red[c] += red[c + s];
        __syncthreads();
    }
    float rn = 1.0f / fmaxf(sqrtf(red[0]), EPS_F);
    mn[n * C_DIM + c]     = (_Float16)(v * rn);
    memT[c * N_SLOTS + n] = (_Float16)v;
}

// ---------------------------------------------------------------------------
// Fused per 32-token tile: normalize -> scores (WMMA) -> softmax (LDS) ->
// attn out (NT store) + recon = attn @ memory (WMMA) -> recon out (NT store)
// ---------------------------------------------------------------------------
__global__ void __launch_bounds__(256)
mb_attn_kernel(const float* __restrict__ z,
               const _Float16* __restrict__ mn,
               const _Float16* __restrict__ memT,
               float* __restrict__ out_recon,
               float* __restrict__ out_attn) {
    __shared__ __align__(16) _Float16 zn_s[MT * C_DIM];     // 16 KB: A tile (f16)
    __shared__ __align__(16) _Float16 at_s[MT * N_SLOTS];   // 32 KB: attn (f16)
    __shared__ __align__(16) float    sc[MT * N_SLOTS];     // 64 KB: staging/scores/recon
    __shared__ float scale_s[MT];
    __shared__ float rowmax[MT];
    __shared__ float rowinv[MT];

    const int tid     = threadIdx.x;           // 256 threads = 8 waves (wave32)
    const int tileIdx = blockIdx.x;            // 2048 tiles
    const int b       = tileIdx >> 7;          // 128 tiles per batch image
    const int tbase   = (tileIdx & 127) * MT;  // contiguous token range

    // ---- stage z tile [MT][C] into sc with padded stride 257 (coalesced, NT) ----
    {
        const float* zb = z + (size_t)b * C_DIM * HW + tbase;
        for (int it = 0; it < 32; ++it) {
            int m = tid & 31;
            int c = it * 8 + (tid >> 5);
            sc[m * 257 + c] = __builtin_nontemporal_load(zb + (size_t)c * HW + m);
        }
    }
    __syncthreads();

    // ---- per-token 1/||z|| (folds in SCALE): 8 lanes per row, shfl reduce ----
    {
        int q = tid & 7, row = tid >> 3;       // 32 rows x 8 lanes
        float s = 0.0f;
        for (int i = 0; i < 32; ++i) {
            float x = sc[row * 257 + q + 8 * i];
            s += x * x;
        }
        for (int off = 1; off < 8; off <<= 1) s += __shfl_xor(s, off, 32);
        if (q == 0) scale_s[row] = SCALE_F / fmaxf(sqrtf(s), EPS_F);
    }
    __syncthreads();

    // ---- convert to f16 A tile ----
    for (int it = 0; it < 32; ++it) {
        zn_s[it * C_DIM + tid] = (_Float16)(sc[it * 257 + tid] * scale_s[it]);
    }
    __syncthreads();

    const int lane = tid & 31;
    const int w    = tid >> 5;            // wave id 0..7
    const int lrow = lane & 15;           // fragment row/col within tile
    const int khi  = (lane >> 4) * 8;     // K sub-offset per ISA A/B layout
    const int mb   = (lane >> 4) * 8;     // D fragment M offset
    const int nn   = lane & 15;           // D fragment N

    // ---- GEMM1: scores[32,512] = zn(f16) @ mn^T(f16), K=256 ----
    // Each wave owns 4 N-tiles; B loaded ONCE and used for both M-tiles
    // (A for both M-tiles preloaded in registers: 16 frags = 128 VGPRs).
    {
        v16h afrag[2][8];
        for (int mt = 0; mt < 2; ++mt)
            for (int ks = 0; ks < 8; ++ks) {
                const _Float16* p = &zn_s[(mt * 16 + lrow) * C_DIM + ks * 32 + khi];
                FragU u; u.h[0] = *(const v8h*)p; u.h[1] = *(const v8h*)(p + 16);
                afrag[mt][ks] = u.v;
            }
        for (int j = 0; j < 4; ++j) {
            int ntile = w * 4 + j;
            v8f acc0 = {0.f,0.f,0.f,0.f,0.f,0.f,0.f,0.f};
            v8f acc1 = {0.f,0.f,0.f,0.f,0.f,0.f,0.f,0.f};
            for (int ks = 0; ks < 8; ++ks) {
                const _Float16* p = mn + (size_t)(ntile * 16 + lrow) * C_DIM + ks * 32 + khi;
                FragU u; u.h[0] = *(const v8h*)p; u.h[1] = *(const v8h*)(p + 16);
                acc0 = __builtin_amdgcn_wmma_f32_16x16x32_f16(
                    false, afrag[0][ks], false, u.v, (short)0, acc0, false, false);
                acc1 = __builtin_amdgcn_wmma_f32_16x16x32_f16(
                    false, afrag[1][ks], false, u.v, (short)0, acc1, false, false);
            }
            for (int r = 0; r < 8; ++r) {
                sc[(mb + r) * N_SLOTS + ntile * 16 + nn]        = acc0[r];
                sc[(16 + mb + r) * N_SLOTS + ntile * 16 + nn]   = acc1[r];
            }
        }
    }
    __syncthreads();

    // ---- softmax over N=512 per row: max & sum via 8-lane shfl reductions ----
    {
        int q = tid & 7, row = tid >> 3;
        float mx = -3.0e38f;
        for (int i = 0; i < 64; ++i) mx = fmaxf(mx, sc[row * N_SLOTS + q + 8 * i]);
        for (int off = 1; off < 8; off <<= 1) mx = fmaxf(mx, __shfl_xor(mx, off, 32));
        float sum = 0.0f;
        for (int i = 0; i < 64; ++i) sum += __expf(sc[row * N_SLOTS + q + 8 * i] - mx);
        for (int off = 1; off < 8; off <<= 1) sum += __shfl_xor(sum, off, 32);
        if (q == 0) { rowmax[row] = mx; rowinv[row] = 1.0f / sum; }
    }
    __syncthreads();

    // ---- attn: coalesced NT f32 global write + f16 LDS copy for GEMM2 ----
    {
        float* attn_out = out_attn + ((size_t)b * HW + tbase) * N_SLOTS;
        for (int it = 0; it < 64; ++it) {
            int j = it * 256 + tid;       // j = row*512 + col
            int r = j >> 9;
            float e = __expf(sc[j] - rowmax[r]) * rowinv[r];
            __builtin_nontemporal_store(e, attn_out + j);
            at_s[j] = (_Float16)e;
        }
    }
    __syncthreads();

    // ---- GEMM2: recon[32,256] = attn(f16) @ memory, K=512 (B = memT rows) ----
    // Each wave owns 2 C-tiles; B loaded ONCE per K-step, used for both M-tiles.
    {
        for (int j = 0; j < 2; ++j) {
            int ntile = w * 2 + j;
            v8f acc0 = {0.f,0.f,0.f,0.f,0.f,0.f,0.f,0.f};
            v8f acc1 = {0.f,0.f,0.f,0.f,0.f,0.f,0.f,0.f};
            for (int ks = 0; ks < 16; ++ks) {
                const _Float16* pb = memT + (size_t)(ntile * 16 + lrow) * N_SLOTS + ks * 32 + khi;
                FragU ub; ub.h[0] = *(const v8h*)pb; ub.h[1] = *(const v8h*)(pb + 16);
                const _Float16* pa0 = &at_s[lrow * N_SLOTS + ks * 32 + khi];
                FragU ua0; ua0.h[0] = *(const v8h*)pa0; ua0.h[1] = *(const v8h*)(pa0 + 16);
                acc0 = __builtin_amdgcn_wmma_f32_16x16x32_f16(
                    false, ua0.v, false, ub.v, (short)0, acc0, false, false);
                const _Float16* pa1 = &at_s[(16 + lrow) * N_SLOTS + ks * 32 + khi];
                FragU ua1; ua1.h[0] = *(const v8h*)pa1; ua1.h[1] = *(const v8h*)(pa1 + 16);
                acc1 = __builtin_amdgcn_wmma_f32_16x16x32_f16(
                    false, ua1.v, false, ub.v, (short)0, acc1, false, false);
            }
            // store transposed [c][m] so final global write is coalesced
            for (int r = 0; r < 8; ++r) {
                sc[(ntile * 16 + nn) * MT + mb + r]      = acc0[r];
                sc[(ntile * 16 + nn) * MT + 16 + mb + r] = acc1[r];
            }
        }
    }
    __syncthreads();

    // ---- recon writeback: per channel, 32 contiguous tokens (NT) ----
    {
        float* rb = out_recon + (size_t)b * C_DIM * HW + tbase;
        for (int it = 0; it < 32; ++it) {
            int m = tid & 31;
            int c = it * 8 + (tid >> 5);
            __builtin_nontemporal_store(sc[c * MT + m], rb + (size_t)c * HW + m);
        }
    }
}

extern "C" void kernel_launch(void* const* d_in, const int* in_sizes, int n_in,
                              void* d_out, int out_size, void* d_ws, size_t ws_size,
                              hipStream_t stream) {
    (void)in_sizes; (void)n_in; (void)out_size; (void)ws_size;
    const float* z      = (const float*)d_in[0];
    const float* memory = (const float*)d_in[1];

    float* out_recon = (float*)d_out;                                  // [B,C,H,W]
    float* out_attn  = out_recon + (size_t)B_DIM * C_DIM * HW;         // [B,HW,N]

    _Float16* mn   = (_Float16*)d_ws;                 // 512*256 f16 = 256 KB
    _Float16* memT = mn + (size_t)N_SLOTS * C_DIM;    // 256*512 f16 = 256 KB

    mb_prep_kernel<<<N_SLOTS, 256, 0, stream>>>(memory, mn, memT);
    mb_attn_kernel<<<(B_DIM * HW) / MT, 256, 0, stream>>>(z, mn, memT, out_recon, out_attn);
}
